// GL_TAGConv_3l_512h_w_k3_44753559224326
// MI455X (gfx1250) — matrix operand order repaired and checked by reading, not compiled, not run on hardware
//
#include <hip/hip_runtime.h>
#include <math.h>

// ---------------------------------------------------------------------------
// TAGConv (3 layers, K=3) for MI455X / gfx1250.
//   - SpMM scatter in f32 (L2-resident gathers + global_atomic_add_f32)
//   - GEMMs in bf16 via v_wmma_f32_16x16x32_bf16, f32 accumulate
//   - Each wave computes a 16x(16*NT) output strip; K-loop is software-
//     pipelined (load kt+1 while WMMAing kt) so loads overlap matrix math.
// ---------------------------------------------------------------------------

typedef __attribute__((ext_vector_type(16))) __bf16 v16bf;
typedef __attribute__((ext_vector_type(8)))  float  v8f;

union BV {
    uint4 u4[2];
    v16bf v;
};

static __device__ __forceinline__ unsigned short f2bf(float f) {
    unsigned int u = __float_as_uint(f);
    unsigned int lsb = (u >> 16) & 1u;
    u += 0x7fffu + lsb;              // round-to-nearest-even
    return (unsigned short)(u >> 16);
}

// ------------------------------- norm pipeline -----------------------------

__global__ void deg_kernel(const long long* __restrict__ col,
                           const float* __restrict__ w,
                           float* __restrict__ deg, long long E) {
    long long e = (long long)blockIdx.x * blockDim.x + threadIdx.x;
    if (e < E) atomicAdd(&deg[col[e]], w[e]);
}

__global__ void dinv_kernel(float* __restrict__ deg, int n) {
    int i = blockIdx.x * blockDim.x + threadIdx.x;
    if (i < n) {
        float d = deg[i];
        deg[i] = d > 0.0f ? rsqrtf(d) : 0.0f;
    }
}

__global__ void norm_kernel(const long long* __restrict__ row,
                            const long long* __restrict__ col,
                            const float* __restrict__ w,
                            const float* __restrict__ dinv,
                            float* __restrict__ nrm, long long E) {
    long long e = (long long)blockIdx.x * blockDim.x + threadIdx.x;
    if (e < E) nrm[e] = dinv[row[e]] * w[e] * dinv[col[e]];
}

// ------------------------------- SpMM scatter ------------------------------
// one thread per (edge, 4 features); gathers hit L2 (h fits in 192MB L2)

__global__ void spmm_scatter(const long long* __restrict__ row,
                             const long long* __restrict__ col,
                             const float* __restrict__ nrm,
                             const float* __restrict__ hin,
                             float* __restrict__ hout,
                             long long E, int F, int shift) {
    long long t = (long long)blockIdx.x * blockDim.x + threadIdx.x;
    long long e = t >> shift;
    if (e >= E) return;
    int f4 = (int)(t & (((long long)1 << shift) - 1)) << 2;
    long long r = row[e], c = col[e];
    float nv = nrm[e];
    const float4 s = *(const float4*)(hin + r * (long long)F + f4);
    float* d = hout + c * (long long)F + f4;
    atomicAdd(d + 0, nv * s.x);
    atomicAdd(d + 1, nv * s.y);
    atomicAdd(d + 2, nv * s.z);
    atomicAdd(d + 3, nv * s.w);
}

// ---------------------------- f32 -> bf16 convert --------------------------

__global__ void cvt_bf16(const float* __restrict__ in,
                         unsigned short* __restrict__ out, long long n4) {
    long long i = (long long)blockIdx.x * blockDim.x + threadIdx.x;
    if (i >= n4) return;
    const float4 v = *(const float4*)(in + i * 4);
    ushort4 o;
    o.x = f2bf(v.x); o.y = f2bf(v.y); o.z = f2bf(v.z); o.w = f2bf(v.w);
    *(ushort4*)(out + i * 4) = o;
}

// --------------------------- weight pre-packing ----------------------------
// Pack W (Fin x Fout, f32 row-major) into per-(K-tile, N-tile) WMMA B layout:
// lane L of tile (kt,nt): VGPR j holds K = kt*32 + (L/16)*8 + 2j (+8 if j>=4) (+1),
// n = nt*16 + L%16. Stored contiguously: 16 bf16 per (tile, lane)
// -> two coalesced b128 loads per tile in the GEMM.

__global__ void pack_w(const float* __restrict__ W,
                       unsigned short* __restrict__ Wp, int Fin, int Fout) {
    int ntN = Fout >> 4;
    int total = (Fin >> 5) * ntN * 32;
    int t = blockIdx.x * blockDim.x + threadIdx.x;
    if (t >= total) return;
    int lane = t & 31;
    int tile = t >> 5;
    int nt = tile % ntN, kt = tile / ntN;
    int n = nt * 16 + (lane & 15);
    int kbase = kt * 32 + (lane >> 4) * 8;
    unsigned short* dst = Wp + (size_t)t * 16;
#pragma unroll
    for (int j = 0; j < 8; ++j) {
        int K = kbase + 2 * j + (j >= 4 ? 8 : 0);
        dst[2 * j]     = f2bf(W[(size_t)K * Fout + n]);
        dst[2 * j + 1] = f2bf(W[(size_t)(K + 1) * Fout + n]);
    }
}

// --------------------------------- GEMM ------------------------------------
// One wave per 16x(16*NT) output strip: one A-tile load feeds NT WMMAs.
// A: 16-bit A-matrix layout (lane half picks K 0..7/16..23 vs 8..15/24..31).
// B: pre-packed (contiguous per n-tile). Software-pipelined: loads for K-step
// kt+1 are issued before the WMMAs of K-step kt, forcing two live load-buffer
// generations (distinct registers) and partial s_wait_loadcnt waits.
// mode: 0 = write (c + bias), 1 = Out += c, 2 = Out = elu(Out + c)

template <int NT>
__global__ __launch_bounds__(256)
void gemm_wmma(const unsigned short* __restrict__ A,
               const unsigned short* __restrict__ Wp,
               const float* __restrict__ bias,
               float* __restrict__ Out,
               int M, int Fin, int Fout, int mode) {
    int ntN = Fout >> 4;
    int ng  = ntN / NT;
    int nTiles = (M >> 4) * ng;
    int wave = (blockIdx.x << 3) + (threadIdx.x >> 5);
    if (wave >= nTiles) return;                 // wave-uniform: EXEC stays all-1s
    int ntg = wave % ng;
    int mt  = wave / ng;
    int lane = threadIdx.x & 31;
    int half = lane >> 4;
    int lq = lane & 15;

    v8f c[NT];
#pragma unroll
    for (int t = 0; t < NT; ++t) c[t] = {};

    const unsigned short* arow  = A + (size_t)(mt * 16 + lq) * Fin + half * 8;
    const unsigned short* bbase = Wp + ((size_t)(ntg * NT) * 32 + lane) * 16;
    size_t bstride = (size_t)ntN * 32 * 16;     // advance one K-tile
    int kts = Fin >> 5;

    // ---- prologue: load K-step 0 ----
    BV a, b[NT];
    a.u4[0] = *(const uint4*)(arow);            // K = half*8 .. half*8+7
    a.u4[1] = *(const uint4*)(arow + 16);       // K = 16+half*8 ..
#pragma unroll
    for (int t = 0; t < NT; ++t) {
        const unsigned short* bt = bbase + (size_t)t * 32 * 16;
        b[t].u4[0] = *(const uint4*)(bt);
        b[t].u4[1] = *(const uint4*)(bt + 8);
    }

    // ---- steady state: load kt+1, compute kt, rotate ----
#pragma unroll 2
    for (int kt = 0; kt < kts - 1; ++kt) {
        BV an, bn[NT];
        const unsigned short* ap = arow + (kt + 1) * 32;
        an.u4[0] = *(const uint4*)(ap);
        an.u4[1] = *(const uint4*)(ap + 16);
        const unsigned short* bp = bbase + (size_t)(kt + 1) * bstride;
#pragma unroll
        for (int t = 0; t < NT; ++t) {
            const unsigned short* bt = bp + (size_t)t * 32 * 16;
            bn[t].u4[0] = *(const uint4*)(bt);
            bn[t].u4[1] = *(const uint4*)(bt + 8);
        }
#pragma unroll
        for (int t = 0; t < NT; ++t) {
            c[t] = __builtin_amdgcn_wmma_f32_16x16x32_bf16(
                       false, a.v, false, b[t].v, (short)0, c[t], false, false);
        }
        a = an;
#pragma unroll
        for (int t = 0; t < NT; ++t) b[t] = bn[t];
    }

    // ---- epilogue: compute last K-step ----
#pragma unroll
    for (int t = 0; t < NT; ++t) {
        c[t] = __builtin_amdgcn_wmma_f32_16x16x32_bf16(
                   false, a.v, false, b[t].v, (short)0, c[t], false, false);
    }

#pragma unroll
    for (int t = 0; t < NT; ++t) {
        int n = (ntg * NT + t) * 16 + lq;
        float bn = (mode == 0) ? bias[n] : 0.0f;
        float* op = Out + (size_t)(mt * 16 + half * 8) * Fout + n;
#pragma unroll
        for (int r = 0; r < 8; ++r) {
            float* p = op + (size_t)r * Fout;
            float v;
            if (mode == 0)      v = c[t][r] + bn;
            else                v = *p + c[t][r];
            if (mode == 2)      v = v > 0.0f ? v : (expf(v) - 1.0f);
            *p = v;
        }
    }
}

// ------------------------------- host driver -------------------------------

namespace {

constexpr int NN = 50000;
constexpr int KHOP = 3;

inline unsigned int blocks_for(long long n, int bs) {
    return (unsigned int)((n + bs - 1) / bs);
}

void launch_gemm(const unsigned short* A, const unsigned short* Wp,
                 const float* bias, float* Out, int Fin, int Fout, int mode,
                 hipStream_t stream) {
    if ((Fout >> 4) % 4 == 0) {
        int nTiles = (NN >> 4) * ((Fout >> 4) / 4);
        gemm_wmma<4><<<(unsigned int)((nTiles + 7) / 8), 256, 0, stream>>>(
            A, Wp, bias, Out, NN, Fin, Fout, mode);
    } else {
        int nTiles = (NN >> 4) * (Fout >> 4);
        gemm_wmma<1><<<(unsigned int)((nTiles + 7) / 8), 256, 0, stream>>>(
            A, Wp, bias, Out, NN, Fin, Fout, mode);
    }
}

void run_layer(const float* in, int Fin, int Fout,
               const unsigned short* wp, const float* bias, float* Out,
               bool elu,
               const long long* row, const long long* col, const float* nrm,
               long long E, float* bufA, float* bufB, unsigned short* hbf,
               hipStream_t stream) {
    const long long nIn = (long long)NN * Fin;
    const int shift = (Fin == 512) ? 7 : 5;     // log2(Fin/4); Fin in {128,512}

    // k = 0: Out = in @ W[0] + b
    cvt_bf16<<<blocks_for(nIn / 4, 256), 256, 0, stream>>>(in, hbf, nIn / 4);
    launch_gemm(hbf, wp, bias, Out, Fin, Fout, 0, stream);

    const float* prev = in;
    for (int k = 1; k <= KHOP; ++k) {
        float* cur = (k & 1) ? bufA : bufB;
        hipMemsetAsync(cur, 0, (size_t)nIn * sizeof(float), stream);
        long long tot = E << shift;
        spmm_scatter<<<blocks_for(tot, 256), 256, 0, stream>>>(
            row, col, nrm, prev, cur, E, Fin, shift);
        cvt_bf16<<<blocks_for(nIn / 4, 256), 256, 0, stream>>>(cur, hbf, nIn / 4);
        int mode = (k == KHOP && elu) ? 2 : 1;
        launch_gemm(hbf, wp + (size_t)k * Fin * Fout, bias, Out,
                    Fin, Fout, mode, stream);
        prev = cur;
    }
}

} // namespace

extern "C" void kernel_launch(void* const* d_in, const int* in_sizes, int n_in,
                              void* d_out, int out_size, void* d_ws, size_t ws_size,
                              hipStream_t stream) {
    const float*     x   = (const float*)d_in[0];
    const long long* ei  = (const long long*)d_in[1];   // (2, E) int64
    const float*     ew  = (const float*)d_in[2];
    const float*     W1  = (const float*)d_in[3];
    const float*     b1  = (const float*)d_in[4];
    const float*     W2  = (const float*)d_in[5];
    const float*     b2  = (const float*)d_in[6];
    const float*     W3  = (const float*)d_in[7];
    const float*     b3  = (const float*)d_in[8];

    const long long E = in_sizes[2];
    const long long* row = ei;
    const long long* col = ei + E;

    // ---- carve workspace ----
    char* ws = (char*)d_ws;
    size_t off = 0;
    auto carve = [&](size_t bytes) -> char* {
        char* p = ws + off;
        off = (off + bytes + 255) & ~(size_t)255;
        return p;
    };
    float*          deg  = (float*)carve((size_t)NN * 4);
    float*          nrm  = (float*)carve((size_t)E * 4);
    unsigned short* wp1  = (unsigned short*)carve((size_t)4 * 128 * 512 * 2);
    unsigned short* wp2  = (unsigned short*)carve((size_t)4 * 512 * 512 * 2);
    unsigned short* wp3  = (unsigned short*)carve((size_t)4 * 512 * 16 * 2);
    float*          bufA = (float*)carve((size_t)NN * 512 * 4);
    float*          bufB = (float*)carve((size_t)NN * 512 * 4);
    float*          O1   = (float*)carve((size_t)NN * 512 * 4);
    float*          O2   = (float*)carve((size_t)NN * 512 * 4);
    unsigned short* hbf  = (unsigned short*)carve((size_t)NN * 512 * 2);

    // ---- edge normalization ----
    hipMemsetAsync(deg, 0, (size_t)NN * sizeof(float), stream);
    deg_kernel<<<blocks_for(E, 256), 256, 0, stream>>>(col, ew, deg, E);
    dinv_kernel<<<blocks_for(NN, 256), 256, 0, stream>>>(deg, NN);
    norm_kernel<<<blocks_for(E, 256), 256, 0, stream>>>(row, col, ew, deg,
                                                        nrm, E);

    // ---- pack weights into WMMA B-layout (bf16) ----
    for (int k = 0; k <= KHOP; ++k) {
        int t1 = (128 >> 5) * (512 >> 4) * 32;
        pack_w<<<blocks_for(t1, 256), 256, 0, stream>>>(
            W1 + (size_t)k * 128 * 512, wp1 + (size_t)k * 128 * 512, 128, 512);
        int t2 = (512 >> 5) * (512 >> 4) * 32;
        pack_w<<<blocks_for(t2, 256), 256, 0, stream>>>(
            W2 + (size_t)k * 512 * 512, wp2 + (size_t)k * 512 * 512, 512, 512);
        int t3 = (512 >> 5) * (16 >> 4) * 32;
        pack_w<<<blocks_for(t3, 256), 256, 0, stream>>>(
            W3 + (size_t)k * 512 * 16, wp3 + (size_t)k * 512 * 16, 512, 16);
    }

    // ---- three TAGConv layers ----
    run_layer(x,  128, 512, wp1, b1, O1, true,
              row, col, nrm, E, bufA, bufB, hbf, stream);
    run_layer(O1, 512, 512, wp2, b2, O2, true,
              row, col, nrm, E, bufA, bufB, hbf, stream);
    run_layer(O2, 512, 16,  wp3, b3, (float*)d_out, false,
              row, col, nrm, E, bufA, bufB, hbf, stream);

    (void)n_in; (void)out_size; (void)ws_size;
}
